// PLRNN_Step_84284438216720
// MI455X (gfx1250) — matrix operand order repaired and checked by reading, not compiled, not run on hardware
//
#include <hip/hip_runtime.h>

#define NB    65536      // batch rows
#define DZ    512
#define MTILE 64
#define LPAD  8
#define LSTR  (DZ + LPAD)   // 520 ushorts per LDS row

typedef __attribute__((ext_vector_type(16))) __bf16       v16bf;
typedef __attribute__((ext_vector_type(8)))  float        v8f;
typedef __attribute__((ext_vector_type(4)))  unsigned int u32x4;

union FragAB { v16bf bf; u32x4 u[2]; };
union Pack8  { unsigned short s[8]; u32x4 v; };

__device__ __forceinline__ unsigned short f32_to_bf16_rne(float f) {
    unsigned int x = __float_as_uint(f);
    unsigned int r = x + 0x7FFFu + ((x >> 16) & 1u);
    return (unsigned short)(r >> 16);
}
__device__ __forceinline__ float bf16f(unsigned short u) {
    return __uint_as_float(((unsigned int)u) << 16);
}

// ---------------------------------------------------------------------------
// Prep: zero-diagonal AW -> bf16 hi/lo in WMMA-B-fragment-packed layout.
// Fragment (kstep, ntile) = 1024 bytes: lane-major, 16 contiguous K bf16/lane.
//   lane = (kk>=16)*16 + (n&15), holds K = kb + (lane>=16 ? 16 : 0) + 0..15
// ---------------------------------------------------------------------------
__global__ void prep_w_kernel(const float* __restrict__ AW,
                              unsigned short* __restrict__ Wh,
                              unsigned short* __restrict__ Wl) {
    int gid = blockIdx.x * blockDim.x + threadIdx.x;   // 0 .. 512*512-1
    int n = gid >> 9;
    int k = gid & 511;
    float v = (n == k) ? 0.0f : AW[n * DZ + k];        // B[k][n] = W[n][k]
    unsigned short hi = f32_to_bf16_rne(v);
    unsigned short lo = f32_to_bf16_rne(v - bf16f(hi));
    int kstep = k >> 5;
    int kk    = k & 31;
    int half  = kk >> 4;
    int klo   = kk & 15;
    int lane  = half * 16 + (n & 15);
    int ntile = n >> 4;
    int idx   = (kstep * 32 + ntile) * 512 + lane * 16 + klo;
    Wh[idx] = hi;
    Wl[idx] = lo;
}

// ---------------------------------------------------------------------------
// Fused layernorm(mean-sub)+ReLU + split-bf16 WMMA GEMM + diag/bias epilogue.
// Block: 256 threads (8 wave32), 64 rows x 512 cols of output.
// ---------------------------------------------------------------------------
__launch_bounds__(256, 1)
__global__ void plrnn_gemm_kernel(const float* __restrict__ z,
                                  const float* __restrict__ AW,
                                  const float* __restrict__ hvec,
                                  const unsigned short* __restrict__ Wh,
                                  const unsigned short* __restrict__ Wl,
                                  float* __restrict__ out) {
    extern __shared__ unsigned short smem[];           // zh[64*520] | zl[64*520]
    unsigned short* zh = smem;
    unsigned short* zl = smem + MTILE * LSTR;

    const int tid  = threadIdx.x;
    const int wave = tid >> 5;
    const int lane = tid & 31;
    const int brow = blockIdx.x * MTILE;

    // ---- Phase 1+2: per-wave rows [8w, 8w+8): mean, relu, bf16 hi/lo -> LDS
    for (int r8 = 0; r8 < 8; ++r8) {
        const int r = wave * 8 + r8;
        const float* zrow = z + (size_t)(brow + r) * DZ;
        const float4* zp = (const float4*)(zrow + lane * 16);
        float vbuf[16];
        float s = 0.0f;
        #pragma unroll
        for (int q = 0; q < 4; ++q) {
            float4 v4 = zp[q];
            vbuf[q * 4 + 0] = v4.x; vbuf[q * 4 + 1] = v4.y;
            vbuf[q * 4 + 2] = v4.z; vbuf[q * 4 + 3] = v4.w;
            s += v4.x + v4.y + v4.z + v4.w;
        }
        #pragma unroll
        for (int off = 16; off > 0; off >>= 1) s += __shfl_xor(s, off, 32);
        const float mean = s * (1.0f / (float)DZ);

        Pack8 ph[2], pl[2];
        #pragma unroll
        for (int i = 0; i < 16; ++i) {
            float a = vbuf[i] - mean;
            a = a > 0.0f ? a : 0.0f;
            unsigned short hi = f32_to_bf16_rne(a);
            unsigned short lo = f32_to_bf16_rne(a - bf16f(hi));
            ph[i >> 3].s[i & 7] = hi;
            pl[i >> 3].s[i & 7] = lo;
        }
        u32x4* dh = (u32x4*)(zh + r * LSTR + lane * 16);
        dh[0] = ph[0].v; dh[1] = ph[1].v;
        u32x4* dl = (u32x4*)(zl + r * LSTR + lane * 16);
        dl[0] = pl[0].v; dl[1] = pl[1].v;
    }
    __syncthreads();

    // ---- Phase 3: WMMA main loop. wave w owns N in [64w, 64w+64).
    v8f acc[4][4] = {};
    const int m   = lane & 15;
    const int kh  = (lane >> 4) << 4;   // 0 or 16 (A-frag K half select)
    const int ntb = wave << 2;          // global n-tile base (16-col tiles)

    #pragma unroll 1
    for (int kstep = 0; kstep < 16; ++kstep) {
        const int kb = kstep << 5;

        FragAB Ah[4], Al[4];
        #pragma unroll
        for (int mt = 0; mt < 4; ++mt) {
            const int base = (mt * 16 + m) * LSTR + kb + kh;
            Ah[mt].u[0] = *(const u32x4*)(zh + base);
            Ah[mt].u[1] = *(const u32x4*)(zh + base + 8);
            Al[mt].u[0] = *(const u32x4*)(zl + base);
            Al[mt].u[1] = *(const u32x4*)(zl + base + 8);
        }

        #pragma unroll
        for (int nt = 0; nt < 4; ++nt) {
            const int fragidx = kstep * 32 + ntb + nt;
            const u32x4* pbh = (const u32x4*)(Wh + fragidx * 512 + lane * 16);
            const u32x4* pbl = (const u32x4*)(Wl + fragidx * 512 + lane * 16);
            FragAB Bh, Bl;
            Bh.u[0] = pbh[0]; Bh.u[1] = pbh[1];
            Bl.u[0] = pbl[0]; Bl.u[1] = pbl[1];
            #pragma unroll
            for (int mt = 0; mt < 4; ++mt) {
                acc[mt][nt] = __builtin_amdgcn_wmma_f32_16x16x32_bf16(
                    false, Ah[mt].bf, false, Bh.bf, (short)0, acc[mt][nt], false, false);
                acc[mt][nt] = __builtin_amdgcn_wmma_f32_16x16x32_bf16(
                    false, Ah[mt].bf, false, Bl.bf, (short)0, acc[mt][nt], false, false);
                acc[mt][nt] = __builtin_amdgcn_wmma_f32_16x16x32_bf16(
                    false, Al[mt].bf, false, Bh.bf, (short)0, acc[mt][nt], false, false);
            }
        }
    }

    // ---- Phase 4: epilogue out = diag(AW)*z + acc + h
    #pragma unroll
    for (int nt = 0; nt < 4; ++nt) {
        const int n  = (ntb + nt) * 16 + m;
        const float An = AW[n * (DZ + 1)];
        const float hn = hvec[n];
        #pragma unroll
        for (int mt = 0; mt < 4; ++mt) {
            const int row0 = brow + mt * 16 + ((lane >> 4) << 3);
            #pragma unroll
            for (int j = 0; j < 8; ++j) {
                const size_t off = (size_t)(row0 + j) * DZ + n;
                out[off] = An * z[off] + acc[mt][nt][j] + hn;
            }
        }
    }
}

extern "C" void kernel_launch(void* const* d_in, const int* in_sizes, int n_in,
                              void* d_out, int out_size, void* d_ws, size_t ws_size,
                              hipStream_t stream) {
    (void)in_sizes; (void)n_in; (void)out_size; (void)ws_size;
    const float* z  = (const float*)d_in[0];
    const float* AW = (const float*)d_in[1];
    const float* h  = (const float*)d_in[2];
    float* out = (float*)d_out;

    unsigned short* Wh = (unsigned short*)d_ws;          // 512 KB
    unsigned short* Wl = Wh + (size_t)DZ * DZ;           // 512 KB

    prep_w_kernel<<<(DZ * DZ) / 256, 256, 0, stream>>>(AW, Wh, Wl);

    const size_t shmem = (size_t)2 * MTILE * LSTR * sizeof(unsigned short); // 133120 B
    plrnn_gemm_kernel<<<NB / MTILE, 256, shmem, stream>>>(z, AW, h, Wh, Wl, out);
}
